// RetNetWInit_57707180589219
// MI455X (gfx1250) — matrix operand order repaired and checked by reading, not compiled, not run on hardware
//
#include <hip/hip_runtime.h>
#include <hip/hip_bf16.h>

// ---------------------------------------------------------------------------
// SSD-style dual conv head on gfx1250 via implicit-GEMM WMMA (bf16 -> f32).
// M = flattened (level, batch, h, w) pixels, N = out channels, K = 256*9.
// K loop = 9 taps x 8 channel chunks; per-tap pointer setup + zero-page pad,
// immediate-offset chunk loads, double-buffered across tap boundaries.
// ---------------------------------------------------------------------------

#define NLEVELS 6
#define P_TOTAL 87360            // sum over levels of 8 * H * W
#define NSTEPS  72               // (256 * 9) / 32
#define ANCH    65520            // total anchors = 6 * 10920

typedef __attribute__((ext_vector_type(16))) __bf16 v16bf;
typedef __attribute__((ext_vector_type(8)))  float  v8f;
typedef unsigned int u32x4 __attribute__((ext_vector_type(4)));

union FragAB { v16bf v; u32x4 q[2]; unsigned short u[16]; };
union FragC  { v8f   v; float f[8]; };

// level tables: pixel offsets, H, W, H*W, log2(H*W), log2(W), anchor offsets
__constant__ int c_poff[7] = {0, 65536, 81920, 86016, 87040, 87296, 87360};
__constant__ int c_lh  [6] = {32, 16, 8, 4, 2, 1};
__constant__ int c_lw  [6] = {256, 128, 64, 32, 16, 8};
__constant__ int c_lhw [6] = {8192, 2048, 512, 128, 32, 8};
__constant__ int c_lsh [6] = {13, 11, 9, 7, 5, 3};     // log2(H*W)
__constant__ int c_lwsh[6] = {8, 7, 6, 5, 4, 3};       // log2(W)
__constant__ int c_aoff[6] = {0, 49152, 61440, 64512, 65280, 65472};

__device__ __forceinline__ unsigned short f2bf(float x) {
  unsigned int u = __float_as_uint(x);
  u += 0x7FFFu + ((u >> 16) & 1u);           // round-to-nearest-even
  return (unsigned short)(u >> 16);
}

__device__ __forceinline__ int levelOf(int p) {
  int l = 0;
#pragma unroll
  for (int i = 1; i < NLEVELS; ++i) l += (p >= c_poff[i]) ? 1 : 0;
  return l;
}

// ---------------------------------------------------------------------------
// Pack fp32 OIHW 3x3 weights into per-lane WMMA B-fragment order (bf16).
// dst[((nt*NSTEPS + s)*32 + lane)*16 + e]; K-order = tap*256 + c,
// B-frag: N = lane&15, K_local = (e&8)*2 + (lane>>4)*8 + (e&7) within the 32-wide step.
// ---------------------------------------------------------------------------
__global__ __launch_bounds__(256) void pack_weights(
    const float* __restrict__ w, unsigned short* __restrict__ dst, int oc_real)
{
  const int tid  = blockIdx.x * 256 + threadIdx.x;
  const int lane = tid & 31;
  const int sg   = tid >> 5;
  const int s    = sg % NSTEPS;
  const int nt   = sg / NSTEPS;
  const int o    = (nt << 4) + (lane & 15);
  const int hk   = lane >> 4;
  const int t    = s >> 3;                  // tap 0..8
  const int cb   = (s & 7) << 5;            // channel base within tap
  const int ky   = t / 3;
  const int kx   = t - ky * 3;

  unsigned short tmp[16];
#pragma unroll
  for (int e = 0; e < 16; ++e) {
    const int c = cb + ((e & 8) << 1) + (hk << 3) + (e & 7);
    float v = 0.0f;
    if (o < oc_real) v = w[(((size_t)o * 256 + c) * 3 + ky) * 3 + kx];
    tmp[e] = f2bf(v);
  }
  u32x4* d = (u32x4*)(dst + (size_t)tid * 16);
  d[0] = ((const u32x4*)tmp)[0];
  d[1] = ((const u32x4*)tmp)[1];
}

// ---------------------------------------------------------------------------
// fp32 NCHW features (6 levels) -> bf16 activations in [pixel][channel].
// ---------------------------------------------------------------------------
__global__ __launch_bounds__(256) void convert_features(
    const float* __restrict__ f0, const float* __restrict__ f1,
    const float* __restrict__ f2, const float* __restrict__ f3,
    const float* __restrict__ f4, const float* __restrict__ f5,
    unsigned short* __restrict__ dst)
{
  const int idx = blockIdx.x * 256 + threadIdx.x;   // < P_TOTAL * 256
  const int p = idx >> 8;
  const int c = idx & 255;
  const int l = levelOf(p);
  const int r = p - c_poff[l];
  const int n = r >> c_lsh[l];
  const int q = r & (c_lhw[l] - 1);
  const float* f = (l == 0) ? f0 : (l == 1) ? f1 : (l == 2) ? f2
                 : (l == 3) ? f3 : (l == 4) ? f4 : f5;
  dst[idx] = f2bf(f[((size_t)n * 256 + c) * (size_t)c_lhw[l] + q]);
}

// zero the padding pixel row (index P_TOTAL) of both activation buffers
__global__ __launch_bounds__(256) void zero_guard(
    unsigned short* __restrict__ g0, unsigned short* __restrict__ g1)
{
  g0[threadIdx.x] = 0;
  g1[threadIdx.x] = 0;
}

// ---------------------------------------------------------------------------
// Per-lane context for the implicit-GEMM loop (all register resident).
// ---------------------------------------------------------------------------
struct LaneCtx {
  const unsigned short* src;
  const unsigned short* pwB[4];   // per n-subtile, lane-adjusted base
  int h[2], w[2], ib[2];          // per m-subtile
  int H, W, half;
};

template<int NSUB>
struct TapPtrs {
  const unsigned short* a[2];     // per m-subtile base (pixel row, lane half)
  const unsigned short* b[NSUB];  // per n-subtile base for this tap
};

// Per-tap setup: neighbor bounds check once; OOB lanes -> zero pixel row.
template<int NSUB>
__device__ __forceinline__ void make_tap(const LaneCtx& cx, int t, int dy, int dx,
                                         TapPtrs<NSUB>& tp)
{
#pragma unroll
  for (int ms = 0; ms < 2; ++ms) {
    const int hh = cx.h[ms] + dy;
    const int ww = cx.w[ms] + dx;
    const bool valid = ((unsigned)hh < (unsigned)cx.H) &
                       ((unsigned)ww < (unsigned)cx.W);
    const int pix = valid ? (cx.ib[ms] + hh * cx.W + ww) : P_TOTAL;
    tp.a[ms] = cx.src + (((size_t)pix) << 8) + (cx.half << 3);
  }
#pragma unroll
  for (int i = 0; i < NSUB; ++i)
    tp.b[i] = cx.pwB[i] + (size_t)t * 4096;    // 8 chunks * 512 ushorts
}

// One 32-wide K chunk: all loads are immediate offsets off the tap pointers.
template<int NSUB>
__device__ __forceinline__ void load_chunk(const TapPtrs<NSUB>& tp, int j,
                                           FragAB* A, FragAB* B)
{
#pragma unroll
  for (int ms = 0; ms < 2; ++ms) {
    const u32x4* ap = (const u32x4*)tp.a[ms];
    A[ms].q[0] = ap[j * 4];          // K_local = half*8 + e      (e = 0..7)
    A[ms].q[1] = ap[j * 4 + 2];      // K_local = 16 + half*8 + e (e = 8..15)
  }
#pragma unroll
  for (int i = 0; i < NSUB; ++i) {
    const u32x4* bp = (const u32x4*)tp.b[i];
    B[i].q[0] = bp[j * 64];
    B[i].q[1] = bp[j * 64 + 1];
  }
}

template<int NSUB>
__device__ __forceinline__ void wmma_step(FragAB* A, FragAB* B, FragC acc[2][4])
{
#pragma unroll
  for (int i = 0; i < NSUB; ++i)
#pragma unroll
    for (int ms = 0; ms < 2; ++ms)
      acc[ms][i].v = __builtin_amdgcn_wmma_f32_16x16x32_bf16(
          false, A[ms].v, false, B[i].v, (short)0, acc[ms][i].v, false, false);
}

// 9 taps x 8 chunks, double-buffered; the first chunk of tap t+1 issues
// before the last WMMAs of tap t (no pipeline drain at tap boundaries).
template<int NSUB>
__device__ __forceinline__ void conv_body(const LaneCtx& cx, FragC acc[2][4])
{
  TapPtrs<NSUB> cur, nxt;
  make_tap<NSUB>(cx, 0, -1, -1, cur);
  FragAB Abuf[2][2];
  FragAB Bbuf[2][NSUB];
  load_chunk<NSUB>(cur, 0, Abuf[0], Bbuf[0]);

  int ky = 0, kx = 0;
  for (int t = 0; t < 9; ++t) {
    int nkx = kx + 1, nky = ky;
    if (nkx == 3) { nkx = 0; ++nky; }
    if (t < 8) make_tap<NSUB>(cx, t + 1, nky - 1, nkx - 1, nxt);
#pragma unroll
    for (int j = 0; j < 8; ++j) {
      FragAB* An = Abuf[(j + 1) & 1];
      FragAB* Bn = Bbuf[(j + 1) & 1];
      if (j < 7)      load_chunk<NSUB>(cur, j + 1, An, Bn);
      else if (t < 8) load_chunk<NSUB>(nxt, 0, An, Bn);
      wmma_step<NSUB>(Abuf[j & 1], Bbuf[j & 1], acc);
    }
    cur = nxt;
    kx = nkx; ky = nky;
  }
}

// ---------------------------------------------------------------------------
// Implicit-GEMM 3x3 SAME conv. Block = 256 threads = 8 waves (wm x wn = 2x4).
// Block tile: 64 pixels x 256 ochans; wave tile: 32 pixels x 64 ochans.
// Mid layers: bias+ReLU -> bf16 acts. Out layers: bias + scatter into d_out
// with the reshape(n, J, 6*H*W) + level-concat index map.
// ---------------------------------------------------------------------------
__global__ __launch_bounds__(256) void conv3x3_wmma(
    const unsigned short* __restrict__ src,   // bf16 [P_TOTAL+1][256]
    const unsigned short* __restrict__ pw,    // packed bf16 weights
    const float* __restrict__ bias,
    unsigned short* __restrict__ dstAct,      // mid-layer destination (or null)
    float* __restrict__ dstOut,               // out-layer destination (or null)
    int n_subtiles, int oc_real, int nJ, int relu)
{
  const int tid  = threadIdx.x;
  const int lane = tid & 31;
  const int wv   = tid >> 5;
  const int wm   = wv & 1;
  const int wn   = wv >> 1;

  // wave-uniform number of active n-subtiles (scalarized -> no EXEC masking)
  int ns = n_subtiles - (wn << 2);
  ns = ns < 0 ? 0 : (ns > 4 ? 4 : ns);
  ns = __builtin_amdgcn_readfirstlane(ns);
  if (ns == 0) return;                         // wave-uniform early exit

  // block never straddles a level (64 | every level size) -> uniform decode
  const int pblk = blockIdx.x * 64;
  const int l    = levelOf(pblk);
  const int W    = c_lw[l];
  const int H    = c_lh[l];
  const int wsh  = c_lwsh[l];
  const int lsh  = c_lsh[l];
  const int hwm  = c_lhw[l] - 1;

  LaneCtx cx;
  cx.src  = src;
  cx.H    = H;
  cx.W    = W;
  cx.half = lane >> 4;
#pragma unroll
  for (int ms = 0; ms < 2; ++ms) {
    const int p = pblk + (((wm << 1) + ms) << 4) + (lane & 15);
    const int r = p - c_poff[l];
    const int n = r >> lsh;
    const int q = r & hwm;
    cx.h[ms]  = q >> wsh;
    cx.w[ms]  = q & (W - 1);
    cx.ib[ms] = c_poff[l] + (n << lsh);
  }
#pragma unroll
  for (int i = 0; i < 4; ++i)
    cx.pwB[i] = pw + ((size_t)(((wn << 2) + i) * NSTEPS) * 32 + (size_t)lane) * 16;

  FragC acc[2][4];
#pragma unroll
  for (int ms = 0; ms < 2; ++ms)
#pragma unroll
    for (int i = 0; i < 4; ++i)
#pragma unroll
      for (int k = 0; k < 8; ++k) acc[ms][i].f[k] = 0.0f;

  if (ns == 4) conv_body<4>(cx, acc);
  else         conv_body<2>(cx, acc);          // only other case is 2 (reg out)

  // -------- epilogue --------
  const int half = lane >> 4;
  if (dstAct) {
#pragma unroll
    for (int ms = 0; ms < 2; ++ms) {
#pragma unroll
      for (int i = 0; i < 4; ++i) {
        const int oc = (((wn << 2) + i) << 4) + (lane & 15);
        const float bv = bias[oc];
#pragma unroll
        for (int k = 0; k < 8; ++k) {
          const int p = pblk + (((wm << 1) + ms) << 4) + (half << 3) + k;
          float v = acc[ms][i].f[k] + bv;
          if (relu) v = v > 0.0f ? v : 0.0f;
          dstAct[((size_t)p << 8) + oc] = f2bf(v);
        }
      }
    }
  } else {
#pragma unroll
    for (int ms = 0; ms < 2; ++ms) {
#pragma unroll
      for (int i = 0; i < 4; ++i) {
        if (i >= ns) continue;
        const int oc = (((wn << 2) + i) << 4) + (lane & 15);
        if (oc >= oc_real) continue;
        const float bv = bias[oc];
        const int j  = oc / 6;          // class/coord index
        const int bb = oc - j * 6;      // box index
#pragma unroll
        for (int k = 0; k < 8; ++k) {
          const int p = pblk + (((wm << 1) + ms) << 4) + (half << 3) + k;
          const int r = p - c_poff[l];
          const int n = r >> lsh;
          const int q = r & hwm;
          const float v = acc[ms][i].f[k] + bv;
          dstOut[(size_t)n * (size_t)nJ * ANCH + (size_t)j * ANCH
                 + c_aoff[l] + (bb << lsh) + q] = v;
        }
      }
    }
  }
}

// ---------------------------------------------------------------------------
extern "C" void kernel_launch(void* const* d_in, const int* in_sizes, int n_in,
                              void* d_out, int out_size, void* d_ws, size_t ws_size,
                              hipStream_t stream)
{
  const float* feat[6];
  for (int i = 0; i < 6; ++i) feat[i] = (const float*)d_in[i];
  const float* reg_w     = (const float*)d_in[6];
  const float* reg_b     = (const float*)d_in[7];
  const float* reg_w_out = (const float*)d_in[8];
  const float* reg_b_out = (const float*)d_in[9];
  const float* cls_w     = (const float*)d_in[10];
  const float* cls_b     = (const float*)d_in[11];
  const float* cls_w_out = (const float*)d_in[12];
  const float* cls_b_out = (const float*)d_in[13];
  float* out = (float*)d_out;

  // workspace (bf16 elements): actA | actB (each + 1 guard row) | packed w
  const size_t SA = (size_t)P_TOTAL * 256 + 256;     // +zero guard pixel
  unsigned short* actA = (unsigned short*)d_ws;
  unsigned short* actB = actA + SA;
  unsigned short* pw0  = actB + SA;
  const size_t PWL = (size_t)16 * NSTEPS * 32 * 16;  // 589824 per 256->256 layer
  unsigned short* pw_reg[4];
  unsigned short* pw_cls[4];
  for (int i = 0; i < 4; ++i) pw_reg[i] = pw0 + (size_t)i * PWL;
  for (int i = 0; i < 4; ++i) pw_cls[i] = pw0 + (size_t)(4 + i) * PWL;
  unsigned short* pw_rout = pw0 + (size_t)8 * PWL;
  unsigned short* pw_cout = pw_rout + (size_t)2 * NSTEPS * 32 * 16;

  dim3 blk(256);

  // --- guard rows + pack all weights to WMMA fragment layout (bf16) ---
  zero_guard<<<1, blk, 0, stream>>>(actA + (size_t)P_TOTAL * 256,
                                    actB + (size_t)P_TOTAL * 256);
  for (int i = 0; i < 4; ++i)
    pack_weights<<<144, blk, 0, stream>>>(reg_w + (size_t)i * 256 * 256 * 9, pw_reg[i], 256);
  for (int i = 0; i < 4; ++i)
    pack_weights<<<144, blk, 0, stream>>>(cls_w + (size_t)i * 256 * 256 * 9, pw_cls[i], 256);
  pack_weights<<<18, blk, 0, stream>>>(reg_w_out, pw_rout, 24);   // 2 subtiles
  pack_weights<<<36, blk, 0, stream>>>(cls_w_out, pw_cout, 54);   // 4 subtiles

  const int convGrid = P_TOTAL / 64;   // 1365 blocks, 64 px x 256 oc each

  // --- regression head: A -> B -> A -> B -> A -> out ---
  convert_features<<<P_TOTAL, blk, 0, stream>>>(feat[0], feat[1], feat[2],
                                                feat[3], feat[4], feat[5], actA);
  conv3x3_wmma<<<convGrid, blk, 0, stream>>>(actA, pw_reg[0], reg_b + 0,   actB, nullptr, 16, 256, 0, 1);
  conv3x3_wmma<<<convGrid, blk, 0, stream>>>(actB, pw_reg[1], reg_b + 256, actA, nullptr, 16, 256, 0, 1);
  conv3x3_wmma<<<convGrid, blk, 0, stream>>>(actA, pw_reg[2], reg_b + 512, actB, nullptr, 16, 256, 0, 1);
  conv3x3_wmma<<<convGrid, blk, 0, stream>>>(actB, pw_reg[3], reg_b + 768, actA, nullptr, 16, 256, 0, 1);
  conv3x3_wmma<<<convGrid, blk, 0, stream>>>(actA, pw_rout, reg_b_out, nullptr, out, 2, 24, 4, 0);

  // --- classification head (re-convert features; actA was consumed) ---
  convert_features<<<P_TOTAL, blk, 0, stream>>>(feat[0], feat[1], feat[2],
                                                feat[3], feat[4], feat[5], actA);
  conv3x3_wmma<<<convGrid, blk, 0, stream>>>(actA, pw_cls[0], cls_b + 0,   actB, nullptr, 16, 256, 0, 1);
  conv3x3_wmma<<<convGrid, blk, 0, stream>>>(actB, pw_cls[1], cls_b + 256, actA, nullptr, 16, 256, 0, 1);
  conv3x3_wmma<<<convGrid, blk, 0, stream>>>(actA, pw_cls[2], cls_b + 512, actB, nullptr, 16, 256, 0, 1);
  conv3x3_wmma<<<convGrid, blk, 0, stream>>>(actB, pw_cls[3], cls_b + 768, actA, nullptr, 16, 256, 0, 1);
  conv3x3_wmma<<<convGrid, blk, 0, stream>>>(actA, pw_cout, cls_b_out, nullptr,
                                             out + (size_t)8 * 4 * ANCH, 4, 54, 9, 0);

  (void)in_sizes; (void)n_in; (void)out_size; (void)ws_size;
}
